// Head_6090263626312
// MI455X (gfx1250) — matrix operand order repaired and checked by reading, not compiled, not run on hardware
//
#include <hip/hip_runtime.h>

// Single causal attention head, flash-attention style, bf16 WMMA with f32 accumulate.
// B=8, T=2048, C=1024, H=128.
// Round 4: harden ASYNC_WAIT (inline-asm s_wait_asynccnt fallback if builtin missing).

#define BSZ 8
#define TSZ 2048
#define CSZ 1024
#define HSZ 128

typedef __attribute__((ext_vector_type(16))) __bf16 v16bf;
typedef __attribute__((ext_vector_type(8)))  float  v8f;
typedef int v4i __attribute__((vector_size(16)));

struct alignas(16) B128 { unsigned u0, u1, u2, u3; };

// LDS strides (elements), padded for bank-conflict-free row-indexed reads
#define KSTR 136   // K tile rows (32 x 128 bf16), +8 pad
#define VSTR 40    // V^T tile rows (128 x 32 bf16), +8 pad
#define PSTR 40    // per-wave P scratch rows (16 x 32 bf16), +8 pad

#if __has_builtin(__builtin_amdgcn_global_load_async_to_lds_b128)
#define HAS_ASYNC_LDS 1
#else
#define HAS_ASYNC_LDS 0
#endif

#if HAS_ASYNC_LDS
#if __has_builtin(__builtin_amdgcn_s_wait_asynccnt)
#define ASYNC_WAIT(n) __builtin_amdgcn_s_wait_asynccnt(n)
#else
#define ASYNC_WAIT(n) asm volatile("s_wait_asynccnt %0" ::"i"(n) : "memory")
#endif
#else
#define ASYNC_WAIT(n)
#endif

__device__ __forceinline__ __bf16 f2bf(float f) { return (__bf16)f; }

// 16B async global->LDS copy (falls back to sync copy if builtin missing)
__device__ __forceinline__ void cp16(__bf16* ldst, const __bf16* gsrc) {
#if HAS_ASYNC_LDS
  __builtin_amdgcn_global_load_async_to_lds_b128(
      (__attribute__((address_space(1))) v4i*)gsrc,
      (__attribute__((address_space(3))) v4i*)ldst, 0, 0);
#else
  *(B128*)ldst = *(const B128*)gsrc;
#endif
}

// 16 contiguous bf16 (32B) -> one B-fragment half-pair
__device__ __forceinline__ v16bf ld32(const __bf16* p) {
  union { B128 b[2]; v16bf v; } t;
  t.b[0] = *(const B128*)(p);
  t.b[1] = *(const B128*)(p + 8);
  return t.v;
}

// A-fragment (16x32 bf16): lanes 0-15 take K{0..15}, lanes 16-31 K{16..31}
__device__ __forceinline__ v16bf ld_af(const __bf16* rowp, int half) {
  const __bf16* p = rowp + half * 16;
  union { B128 b[2]; v16bf v; } t;
  t.b[0] = *(const B128*)(p);
  t.b[1] = *(const B128*)(p + 8);
  return t.v;
}

__device__ __forceinline__ v8f wmma_bf16(v16bf a, v16bf b, v8f c) {
  return __builtin_amdgcn_wmma_f32_16x16x32_bf16(false, a, false, b, (short)0, c,
                                                 false, false);
}

// ---------------- kernel 0: weights -> bf16, transposed [H][C] ----------------
__global__ void __launch_bounds__(256)
wt_kernel(const float* __restrict__ Wq, const float* __restrict__ Wk,
          const float* __restrict__ Wv, __bf16* __restrict__ wqt,
          __bf16* __restrict__ wkt, __bf16* __restrict__ wvt) {
  int idx = blockIdx.x * 256 + threadIdx.x;   // 0 .. H*C-1
  int h = idx >> 10;
  int c = idx & (CSZ - 1);
  wqt[h * CSZ + c] = f2bf(Wq[c * HSZ + h]);
  wkt[h * CSZ + c] = f2bf(Wk[c * HSZ + h]);
  wvt[h * CSZ + c] = f2bf(Wv[c * HSZ + h]);
}

// ---------------- kernel 1: projections q,k,v (WMMA) ----------------
// block = 256 thr = 8 waves; block handles 16 rows of x; wave w -> cols w*16..+15
__global__ void __launch_bounds__(256)
proj_kernel(const float* __restrict__ x, const __bf16* __restrict__ wqt,
            const __bf16* __restrict__ wkt, const __bf16* __restrict__ wvt,
            __bf16* __restrict__ qb, __bf16* __restrict__ kb,
            __bf16* __restrict__ vtb) {
  __shared__ alignas(16) float ldsX[16 * 36];   // 16x32 f32, +4 pad
  const int tid  = threadIdx.x;
  const int wave = tid >> 5, lane = tid & 31;
  const int half = lane >> 4, l16 = lane & 15;
  const int rowBase = blockIdx.x * 16;
  const int hcol = wave * 16 + l16;

  v8f cq = {}, ck = {}, cv = {};

  for (int kc = 0; kc < CSZ; kc += 32) {
    {  // cooperative stage: 512 floats, 2 per thread
      int r = tid >> 4, c = (tid & 15) * 2;
      const float2 s = *(const float2*)&x[(size_t)(rowBase + r) * CSZ + kc + c];
      ldsX[r * 36 + c]     = s.x;
      ldsX[r * 36 + c + 1] = s.y;
    }
    __syncthreads();

    union { __bf16 h[16]; v16bf v; } af;
    {
      const float* rp = &ldsX[l16 * 36 + half * 16];
#pragma unroll
      for (int j = 0; j < 8; ++j) {
        af.h[j]     = f2bf(rp[j]);
        af.h[8 + j] = f2bf(rp[8 + j]);
      }
    }
    const int wofs = hcol * CSZ + kc + half * 16;
    v16bf bq = ld32(&wqt[wofs]);
    v16bf bk = ld32(&wkt[wofs]);
    v16bf bv = ld32(&wvt[wofs]);
    cq = wmma_bf16(af.v, bq, cq);
    ck = wmma_bf16(af.v, bk, ck);
    cv = wmma_bf16(af.v, bv, cv);
    __syncthreads();
  }

  // Q,K row-major [B*T][H] bf16
#pragma unroll
  for (int r = 0; r < 8; ++r) {
    int row = rowBase + r + 8 * half;
    qb[(size_t)row * HSZ + hcol] = f2bf(cq[r]);
    kb[(size_t)row * HSZ + hcol] = f2bf(ck[r]);
  }
  // V transposed [B][H][T] bf16 -> contiguous 8-elem pack per lane
  union { __bf16 h[8]; B128 b; } pk;
#pragma unroll
  for (int r = 0; r < 8; ++r) pk.h[r] = f2bf(cv[r]);
  int bb = rowBase >> 11;                    // row / T
  int t0 = (rowBase & (TSZ - 1)) + 8 * half;
  *(B128*)&vtb[((size_t)bb * HSZ + hcol) * TSZ + t0] = pk.b;
}

// ---------------- kernel 2: flash attention ----------------
// block = (b, 128-query tile); wave w -> query rows qBase + w*16 .. +15
// Double-buffered async staging of K/V^T tiles (4 async B128 per thread per tile).
__global__ void __launch_bounds__(256)
attn_kernel(const __bf16* __restrict__ qb, const __bf16* __restrict__ kbm,
            const __bf16* __restrict__ vtb, float* __restrict__ out) {
  __shared__ alignas(16) __bf16 ldsK[2][32 * KSTR];
  __shared__ alignas(16) __bf16 ldsV[2][128 * VSTR];
  __shared__ alignas(16) __bf16 ldsP[8 * 16 * PSTR];

  const int tid  = threadIdx.x;
  const int wave = tid >> 5, lane = tid & 31;
  const int half = lane >> 4, l16 = lane & 15;
  const int b     = blockIdx.x >> 4;          // 16 q-tiles per batch
  const int qBase = (blockIdx.x & 15) * 128;
  const int qw    = qBase + wave * 16;
  __bf16* pScr = &ldsP[wave * 16 * PSTR];

  // per-thread staging coordinates (fixed across tiles)
  const int kr = tid >> 3, ksE = (tid & 7) * 16;       // K tile: 32 rows x 128
  const int vh = tid >> 1, vsE = (tid & 1) * 16;       // V^T tile: 128 rows x 32
  const __bf16* kSrcRow = kbm + ((size_t)b * TSZ + kr) * HSZ + ksE;
  const __bf16* vSrcRow = vtb + ((size_t)b * HSZ + vh) * TSZ + vsE;

  auto stage = [&](int kt, int buf) {
    const int keyBase = kt << 5;
    const __bf16* ks = kSrcRow + (size_t)keyBase * HSZ;
    __bf16* kd = &ldsK[buf][kr * KSTR + ksE];
    cp16(kd, ks);
    cp16(kd + 8, ks + 8);
    const __bf16* vs = vSrcRow + keyBase;
    __bf16* vd = &ldsV[buf][vh * VSTR + vsE];
    cp16(vd, vs);
    cp16(vd + 8, vs + 8);
  };

  // q A-fragments (row M = l16 for both halves), 4 K-steps cover H=128
  const __bf16* qrow = qb + ((size_t)b * TSZ + qw + l16) * HSZ;
  v16bf qa[4];
#pragma unroll
  for (int f = 0; f < 4; ++f) qa[f] = ld_af(qrow + 32 * f, half);

  v8f acc[8] = {};
  float mrow[8], lrow[8];
#pragma unroll
  for (int r = 0; r < 8; ++r) { mrow[r] = -__builtin_inff(); lrow[r] = 0.0f; }

  const int kbMax = (qBase + 128) >> 5;
  stage(0, 0);
  for (int kt = 0; kt < kbMax; ++kt) {
    if (kt + 1 < kbMax) {
      stage(kt + 1, (kt + 1) & 1);
      ASYNC_WAIT(4);   // previous tile's 4 async ops complete; next 4 in flight
    } else {
      ASYNC_WAIT(0);
    }
    __syncthreads();

    const __bf16* K = ldsK[kt & 1];
    const __bf16* V = ldsV[kt & 1];
    const int keyBase = kt << 5;

    if (keyBase <= qw + 15) {  // wave-uniform causal skip
      // scores: two 16x16 C tiles covering keys keyBase..+31
      v8f s0 = {}, s1 = {};
#pragma unroll
      for (int hs = 0; hs < 4; ++hs) {
        v16bf b0 = ld32(&K[l16 * KSTR + hs * 32 + half * 16]);
        v16bf b1 = ld32(&K[(l16 + 16) * KSTR + hs * 32 + half * 16]);
        s0 = wmma_bf16(qa[hs], b0, s0);
        s1 = wmma_bf16(qa[hs], b1, s1);
      }
      const float scale = 0.08838834764831845f;  // 1/sqrt(128)
      const bool needMask = (keyBase + 31) > qw;
      const int key0 = keyBase + l16, key1 = key0 + 16;
#pragma unroll
      for (int r = 0; r < 8; ++r) {
        float v0 = s0[r] * scale, v1 = s1[r] * scale;
        if (needMask) {
          int rowI = qw + r + 8 * half;
          v0 = (key0 > rowI) ? -__builtin_inff() : v0;
          v1 = (key1 > rowI) ? -__builtin_inff() : v1;
        }
        // row max over 16 lanes of this half
        float rm = fmaxf(v0, v1);
#pragma unroll
        for (int off = 8; off; off >>= 1) rm = fmaxf(rm, __shfl_xor(rm, off));
        float mn  = fmaxf(mrow[r], rm);
        bool vld  = (mn > -__builtin_inff());
        float mnS = vld ? mn : 0.0f;
        float corr = vld ? __expf(mrow[r] - mnS) : 1.0f;
        float p0 = vld ? __expf(v0 - mnS) : 0.0f;
        float p1 = vld ? __expf(v1 - mnS) : 0.0f;
        float rs = p0 + p1;
#pragma unroll
        for (int off = 8; off; off >>= 1) rs += __shfl_xor(rs, off);
        lrow[r] = lrow[r] * corr + rs;
        mrow[r] = mn;
#pragma unroll
        for (int ht = 0; ht < 8; ++ht) acc[ht][r] *= corr;
        int M = r + 8 * half;  // C-layout -> P scratch row-major [M][32]
        pScr[M * PSTR + l16]      = f2bf(p0);
        pScr[M * PSTR + 16 + l16] = f2bf(p1);
      }
      // P(16x32) x V(32x128): re-load P in A layout from per-wave scratch
      v16bf pa = ld_af(&pScr[l16 * PSTR], half);
#pragma unroll
      for (int ht = 0; ht < 8; ++ht) {
        v16bf vb = ld32(&V[(ht * 16 + l16) * VSTR + half * 16]);
        acc[ht] = wmma_bf16(pa, vb, acc[ht]);
      }
    }
    __syncthreads();  // all waves done with buf[kt&1] before it is re-staged
  }

  // normalize and store fp32 [B][T][H]
#pragma unroll
  for (int r = 0; r < 8; ++r) {
    float inv = 1.0f / lrow[r];
    int rowI = qw + r + 8 * half;
    float* orow = out + ((size_t)b * TSZ + rowI) * HSZ;
#pragma unroll
    for (int ht = 0; ht < 8; ++ht) orow[ht * 16 + l16] = acc[ht][r] * inv;
  }
}

extern "C" void kernel_launch(void* const* d_in, const int* in_sizes, int n_in,
                              void* d_out, int out_size, void* d_ws, size_t ws_size,
                              hipStream_t stream) {
  (void)in_sizes; (void)n_in; (void)out_size; (void)ws_size;
  const float* x  = (const float*)d_in[0];
  const float* Wq = (const float*)d_in[1];
  const float* Wk = (const float*)d_in[2];
  const float* Wv = (const float*)d_in[3];
  float* out = (float*)d_out;

  __bf16* p = (__bf16*)d_ws;
  const size_t NR = (size_t)BSZ * TSZ;      // 16384 rows
  __bf16* qb  = p; p += NR * HSZ;           // 4 MB
  __bf16* kb  = p; p += NR * HSZ;           // 4 MB
  __bf16* vtb = p; p += NR * HSZ;           // 4 MB ([B][H][T])
  __bf16* wqt = p; p += (size_t)HSZ * CSZ;  // 256 KB each
  __bf16* wkt = p; p += (size_t)HSZ * CSZ;
  __bf16* wvt = p; p += (size_t)HSZ * CSZ;

  wt_kernel<<<dim3((HSZ * CSZ) / 256), dim3(256), 0, stream>>>(Wq, Wk, Wv, wqt, wkt, wvt);
  proj_kernel<<<dim3((unsigned)(NR / 16)), dim3(256), 0, stream>>>(x, wqt, wkt, wvt,
                                                                   qb, kb, vtb);
  attn_kernel<<<dim3(BSZ * (TSZ / 128)), dim3(256), 0, stream>>>(qb, kb, vtb, out);
}
